// GAT_91036126806755
// MI455X (gfx1250) — compile-verified, matrix operand order checked by reading, not run on hardware
//
#include <hip/hip_runtime.h>

typedef __attribute__((ext_vector_type(16))) _Float16 v16h;
typedef __attribute__((ext_vector_type(4)))  _Float16 v4h;
typedef __attribute__((ext_vector_type(8)))  float    v8f;

#define BT      48          // B*T = 4*12
#define NTOK    1024        // N
#define FDIM    512         // F
#define M_TOTAL (BT * NTOK) // 49152 rows for the QKV projection GEMM

// ---------------------------------------------------------------------------
// Fragment loader: CDNA5 16-bit A-matrix 16x32 layout (ISA 7.12.2).
// lane L: row = L%16 ; halfs 0..7  = K[(L/16)*8 .. +7]
//                      halfs 8..15 = K[(L/16)*8+16 .. +23]
// p must already point at element [row][kbase]. Two b128 loads per lane.
// B-operands (= A-layout of B^T) load identically from row-major B^T.
// ---------------------------------------------------------------------------
__device__ __forceinline__ v16h load_frag(const _Float16* p) {
    v16h r;
    reinterpret_cast<float4*>(&r)[0] = *reinterpret_cast<const float4*>(p);
    reinterpret_cast<float4*>(&r)[1] = *reinterpret_cast<const float4*>(p + 16);
    return r;
}

// ---------------------------------------------------------------------------
// Kernel 0a: X (f32) -> Xh (f16), vectorized
// ---------------------------------------------------------------------------
__global__ __launch_bounds__(256) void cvt_x_kernel(const float* __restrict__ x,
                                                    _Float16* __restrict__ xh,
                                                    int n4) {
    int i = blockIdx.x * blockDim.x + threadIdx.x;
    if (i < n4) {
        float4 v = reinterpret_cast<const float4*>(x)[i];
        v4h h = { (_Float16)v.x, (_Float16)v.y, (_Float16)v.z, (_Float16)v.w };
        reinterpret_cast<v4h*>(xh)[i] = h;
    }
}

// ---------------------------------------------------------------------------
// Kernel 0b: W[f][g] (f32) -> Wt[g][f] (f16) for all three weight matrices
// ---------------------------------------------------------------------------
__global__ __launch_bounds__(256) void cvt_w_kernel(const float* __restrict__ Wq,
                                                    const float* __restrict__ Wk,
                                                    const float* __restrict__ Wv,
                                                    _Float16* __restrict__ Wtq,
                                                    _Float16* __restrict__ Wtk,
                                                    _Float16* __restrict__ Wtv) {
    int i = blockIdx.x * blockDim.x + threadIdx.x;   // i = g*512 + f
    if (i < FDIM * FDIM) {
        int g = i >> 9;
        int f = i & (FDIM - 1);
        int src = f * FDIM + g;
        Wtq[i] = (_Float16)Wq[src];
        Wtk[i] = (_Float16)Wk[src];
        Wtv[i] = (_Float16)Wv[src];
    }
}

// ---------------------------------------------------------------------------
// Kernel A: fused QKV projection GEMM.
//   Xh[49152 x 512] x Wt{q,k,v}^T -> Qh,Kh (f16) and Vf (f32), + bias.
// Grid: (M/64, 512/64). Block: 256 threads = 8 waves.
// Wave (r=wid&3, c=wid>>2) owns a 16x32 output region = 2 WMMA tiles,
// accumulated for all 3 weight matrices (X fragment reused 6x).
// ---------------------------------------------------------------------------
__global__ __launch_bounds__(256) void qkv_gemm_kernel(
    const _Float16* __restrict__ Xh,
    const _Float16* __restrict__ Wtq,
    const _Float16* __restrict__ Wtk,
    const _Float16* __restrict__ Wtv,
    const float* __restrict__ bq,
    const float* __restrict__ bk,
    const float* __restrict__ bv,
    _Float16* __restrict__ Qh,
    _Float16* __restrict__ Kh,
    float* __restrict__ Vf) {

    const int lane    = threadIdx.x & 31;
    const int wid     = threadIdx.x >> 5;
    const int rowBase = blockIdx.x * 64 + (wid & 3) * 16;
    const int colBase = blockIdx.y * 64 + (wid >> 2) * 32;

    const int r15  = lane & 15;
    const int kSub = (lane >> 4) * 8;

    const v8f vzero = {0.f, 0.f, 0.f, 0.f, 0.f, 0.f, 0.f, 0.f};
    v8f acc[3][2];
#pragma unroll
    for (int i = 0; i < 3; ++i)
#pragma unroll
        for (int j = 0; j < 2; ++j) acc[i][j] = vzero;

    const _Float16* aPtr = Xh + (size_t)(rowBase + r15) * FDIM + kSub;

#pragma unroll 4
    for (int k = 0; k < FDIM; k += 32) {
        v16h a = load_frag(aPtr + k);
#pragma unroll
        for (int sub = 0; sub < 2; ++sub) {
            size_t boff = (size_t)(colBase + sub * 16 + r15) * FDIM + kSub + k;
            v16h bQ = load_frag(Wtq + boff);
            v16h bK = load_frag(Wtk + boff);
            v16h bV = load_frag(Wtv + boff);
            acc[0][sub] = __builtin_amdgcn_wmma_f32_16x16x32_f16(
                false, a, false, bQ, (short)0, acc[0][sub], false, false);
            acc[1][sub] = __builtin_amdgcn_wmma_f32_16x16x32_f16(
                false, a, false, bK, (short)0, acc[1][sub], false, false);
            acc[2][sub] = __builtin_amdgcn_wmma_f32_16x16x32_f16(
                false, a, false, bV, (short)0, acc[2][sub], false, false);
        }
    }

    // C/D layout (ISA 7.12.2): VGPR r, lanes 0-15 -> M=r, N=lane;
    //                                  lanes 16-31 -> M=r+8, N=lane-16.
    const int mhi = (lane >> 4) * 8;
#pragma unroll
    for (int sub = 0; sub < 2; ++sub) {
        const int col = colBase + sub * 16 + r15;
        const float biasQ = bq[col];
        const float biasK = bk[col];
        const float biasV = bv[col];
#pragma unroll
        for (int r = 0; r < 8; ++r) {
            const size_t o = (size_t)(rowBase + r + mhi) * FDIM + col;
            Qh[o] = (_Float16)(acc[0][sub][r] + biasQ);
            Kh[o] = (_Float16)(acc[1][sub][r] + biasK);
            Vf[o] = acc[2][sub][r] + biasV;
        }
    }
}

// ---------------------------------------------------------------------------
// Kernel B: per (bt, 16-row stripe) compute
//   diagE[n]  = exp(Q[n].K[n] / sqrt(F))
//   rowsum[n] = sum_m exp(Q[n].K[m] / sqrt(F))
// (max-subtraction cancels exactly in diag/rowsum, so unstabilized exp is
//  mathematically identical to softmax; scores are O(1) here.)
// Q stripe DMA'd into LDS with GLOBAL_LOAD_ASYNC_TO_LDS_B128 (ASYNCcnt path,
// no VGPR round-trip); wave w owns column tiles c === w (mod 8).
// ---------------------------------------------------------------------------
__global__ __launch_bounds__(256) void attn_diag_kernel(
    const _Float16* __restrict__ Qh,
    const _Float16* __restrict__ Kh,
    float* __restrict__ diagE,
    float* __restrict__ rowsum) {

    __shared__ _Float16 Qs[16 * FDIM];   // 16 KB
    __shared__ float rsLDS[16];

    const int bt   = blockIdx.y;
    const int s    = blockIdx.x;          // row stripe 0..63
    const int lane = threadIdx.x & 31;
    const int w    = threadIdx.x >> 5;

    // --- async DMA of the Q stripe (16 rows x 512 halfs = 16 KB) into LDS ---
    // Each thread issues four 16B global->LDS copies; tracked by ASYNCcnt.
    {
        const char* src = reinterpret_cast<const char*>(
            Qh + ((size_t)bt * NTOK + s * 16) * FDIM);
        unsigned ldsBase = (unsigned)(size_t)(&Qs[0]);
#pragma unroll
        for (int i = 0; i < 4; ++i) {
            unsigned off      = threadIdx.x * 16u + (unsigned)i * 4096u;
            unsigned ldsAddr  = ldsBase + off;
            const void* gaddr = src + off;
            asm volatile("global_load_async_to_lds_b128 %0, %1, off"
                         :
                         : "v"(ldsAddr), "v"(gaddr)
                         : "memory");
        }
        asm volatile("s_wait_asynccnt 0x0" ::: "memory");
    }
    if (threadIdx.x < 16) rsLDS[threadIdx.x] = 0.f;
    __syncthreads();

    const int r15  = lane & 15;
    const int kSub = (lane >> 4) * 8;
    const float inv_sqrt_f = 0.04419417382415922f;  // 1/sqrt(512)

    const _Float16* qs = Qs + r15 * FDIM + kSub;

    float rs[8] = {0.f, 0.f, 0.f, 0.f, 0.f, 0.f, 0.f, 0.f};

    for (int j = 0; j < 8; ++j) {
        const int c = w + 8 * j;   // column tile 0..63
        v8f acc = {0.f, 0.f, 0.f, 0.f, 0.f, 0.f, 0.f, 0.f};
        const _Float16* kg =
            Kh + ((size_t)bt * NTOK + c * 16 + r15) * FDIM + kSub;
#pragma unroll 4
        for (int k = 0; k < FDIM; k += 32) {
            v16h a = load_frag(qs + k);        // ds reads
            v16h b = load_frag(kg + k);        // global reads
            acc = __builtin_amdgcn_wmma_f32_16x16x32_f16(
                false, a, false, b, (short)0, acc, false, false);
        }
#pragma unroll
        for (int r = 0; r < 8; ++r) {
            float e = __expf(acc[r] * inv_sqrt_f);
            rs[r] += e;
            // diagonal element (i,i): i<8 -> VGPR i, lane i;
            //                         i>=8 -> VGPR i-8, lane 24+(i-8)
            if (c == s && (lane == r || lane == r + 24)) {
                int i = (lane < 16) ? r : r + 8;
                diagE[(size_t)bt * NTOK + s * 16 + i] = e;
            }
        }
    }

    // reduce over the 16 lanes of each half-wave (wave32 butterflies)
#pragma unroll
    for (int r = 0; r < 8; ++r) {
        float v = rs[r];
        v += __shfl_xor(v, 1, 32);
        v += __shfl_xor(v, 2, 32);
        v += __shfl_xor(v, 4, 32);
        v += __shfl_xor(v, 8, 32);
        if (lane == 0)  atomicAdd(&rsLDS[r], v);       // row s*16+r
        if (lane == 16) atomicAdd(&rsLDS[r + 8], v);   // row s*16+r+8
    }
    __syncthreads();
    if (threadIdx.x < 16)
        rowsum[(size_t)bt * NTOK + s * 16 + threadIdx.x] = rsLDS[threadIdx.x];
}

// ---------------------------------------------------------------------------
// Kernel C: out[n][f] = (diagE[n]/rowsum[n]) * V[n][f], float4-vectorized
// ---------------------------------------------------------------------------
__global__ __launch_bounds__(256) void finalize_kernel(
    const float* __restrict__ Vf,
    const float* __restrict__ diagE,
    const float* __restrict__ rowsum,
    float* __restrict__ out, int n4) {
    int i = blockIdx.x * blockDim.x + threadIdx.x;
    if (i < n4) {
        int row = i >> 7;                    // (i*4)/512
        float sc = diagE[row] / rowsum[row];
        float4 v = reinterpret_cast<const float4*>(Vf)[i];
        v.x *= sc; v.y *= sc; v.z *= sc; v.w *= sc;
        reinterpret_cast<float4*>(out)[i] = v;
    }
}

// ---------------------------------------------------------------------------
extern "C" void kernel_launch(void* const* d_in, const int* in_sizes, int n_in,
                              void* d_out, int out_size, void* d_ws, size_t ws_size,
                              hipStream_t stream) {
    const float* x  = (const float*)d_in[0];
    const float* Wq = (const float*)d_in[1];
    const float* bq = (const float*)d_in[2];
    const float* Wk = (const float*)d_in[3];
    const float* bk = (const float*)d_in[4];
    const float* Wv = (const float*)d_in[5];
    const float* bv = (const float*)d_in[6];
    float* out = (float*)d_out;

    const size_t nElem = (size_t)M_TOTAL * FDIM;          // 25,165,824
    char* ws = (char*)d_ws;
    size_t off = 0;
    auto carve = [&](size_t bytes) {
        char* p = ws + off;
        off += (bytes + 255) & ~(size_t)255;
        return p;
    };
    _Float16* Xh  = (_Float16*)carve(nElem * 2);          // 50 MB
    _Float16* Wtq = (_Float16*)carve(FDIM * FDIM * 2);
    _Float16* Wtk = (_Float16*)carve(FDIM * FDIM * 2);
    _Float16* Wtv = (_Float16*)carve(FDIM * FDIM * 2);
    _Float16* Qh  = (_Float16*)carve(nElem * 2);          // 50 MB
    _Float16* Kh  = (_Float16*)carve(nElem * 2);          // 50 MB
    float*    Vf  = (float*)carve(nElem * 4);             // 100 MB
    float*    dE  = (float*)carve((size_t)M_TOTAL * 4);
    float*    rS  = (float*)carve((size_t)M_TOTAL * 4);
    (void)ws_size; (void)in_sizes; (void)n_in; (void)out_size;

    // 0) convert / transpose to f16
    {
        int n4 = (int)(nElem / 4);
        cvt_x_kernel<<<(n4 + 255) / 256, 256, 0, stream>>>(x, Xh, n4);
        cvt_w_kernel<<<(FDIM * FDIM + 255) / 256, 256, 0, stream>>>(
            Wq, Wk, Wv, Wtq, Wtk, Wtv);
    }
    // A) fused QKV projection (WMMA f16 -> f32)
    {
        dim3 grid(M_TOTAL / 64, FDIM / 64);   // (768, 8)
        qkv_gemm_kernel<<<grid, 256, 0, stream>>>(Xh, Wtq, Wtk, Wtv,
                                                  bq, bk, bv, Qh, Kh, Vf);
    }
    // B) diagonal + softmax row sums (WMMA + exp + wave32 reduce)
    {
        dim3 grid(NTOK / 16, BT);             // (64, 48)
        attn_diag_kernel<<<grid, 256, 0, stream>>>(Qh, Kh, dE, rS);
    }
    // C) scale V
    {
        int n4 = (int)(nElem / 4);
        finalize_kernel<<<(n4 + 255) / 256, 256, 0, stream>>>(Vf, dE, rS, out, n4);
    }
}
